// GlobalSpatialAttention_72825465471219
// MI455X (gfx1250) — matrix operand order repaired and checked by reading, not compiled, not run on hardware
//
#include <hip/hip_runtime.h>
#include <hip/hip_bf16.h>

typedef __bf16 bf16_t;
typedef __attribute__((ext_vector_type(16))) __bf16 v16bf;
typedef __attribute__((ext_vector_type(8)))  float  v8f;

// ---------- bf16 helpers ----------
__device__ __forceinline__ bf16_t f2bf(float f) {
    unsigned u = __float_as_uint(f);
    u += 0x7fffu + ((u >> 16) & 1u);          // round-to-nearest-even
    return __builtin_bit_cast(__bf16, (unsigned short)(u >> 16));
}
__device__ __forceinline__ float bf2f(bf16_t b) {
    unsigned u = ((unsigned)__builtin_bit_cast(unsigned short, b)) << 16;
    return __uint_as_float(u);
}
// monotone float<->unsigned mapping for atomicMax on floats
__device__ __forceinline__ unsigned flipf(float f) {
    unsigned u = __float_as_uint(f);
    return (u & 0x80000000u) ? ~u : (u | 0x80000000u);
}
__device__ __forceinline__ float unflipf(unsigned e) {
    unsigned u = (e & 0x80000000u) ? (e ^ 0x80000000u) : ~e;
    return __uint_as_float(u);
}

// ---------- WMMA fragment loaders (CDNA5 16-bit layouts, wave32) ----------
// A 16x32 bf16, row-major over K: lane m = lane&15, half = lane>>4.
// elements 0..15 -> K = 16*half+0..15
__device__ __forceinline__ v16bf load_a_rm(const bf16_t* rowptr, int half) {
    union { uint4 u4[2]; v16bf v; } u;
    u.u4[0] = *(const uint4*)(rowptr + 16 * half);
    u.u4[1] = *(const uint4*)(rowptr + 16 * half + 8);
    return u.v;
}
// B 32x16 bf16 from K-contiguous (transposed) storage: lane col = lane&15,
// halfK = lane>>4, elements 0..15 -> K = 16*halfK + 0..15
__device__ __forceinline__ v16bf load_b_tr(const bf16_t* rowptr, int halfK) {
    union { uint4 u4[2]; v16bf v; } u;
    const uint4* p = (const uint4*)(rowptr + 16 * halfK);
    u.u4[0] = p[0];
    u.u4[1] = p[1];
    return u.v;
}

#define WMMA_BF16(A, B, C) \
    __builtin_amdgcn_wmma_f32_16x16x32_bf16(false, (A), false, (B), (short)0, (C), false, false)

// ---------- kernels ----------
__global__ void cvt_f32_bf16_kernel(const float* __restrict__ in, bf16_t* __restrict__ out, int n) {
    int i = blockIdx.x * 256 + threadIdx.x;
    if (i < n) out[i] = f2bf(in[i]);
}

__global__ void init_stats_kernel(unsigned* dmax, float* dsum) {
    int i = threadIdx.x;
    if (i < 8) { dmax[i] = 0u; dsum[i] = 0.0f; }
}

// Implicit-GEMM conv via WMMA. x: (8,256,64,64) bf16, w: (64, 256*KH*KW) bf16.
// One wave computes the full Cr=64 column strip for its 16 spatial positions:
// the (expensive) im2col B gather is done once per K-chunk and feeds 4 weight
// fragments / 4 accumulators.
// TRANS=true  -> store out[(n*4096+p)*64 + c]   (HW-major, for q/k)
// TRANS=false -> store out[(n*64+c)*4096 + p]   (C-major, for v)
template <int KH, int KW, int PAD, bool TRANS>
__global__ void conv_wmma_kernel(const bf16_t* __restrict__ x, const bf16_t* __restrict__ w,
                                 const float* __restrict__ bias, bf16_t* __restrict__ out) {
    constexpr int KTOT = 256 * KH * KW;
    const int lane = threadIdx.x & 31, wave = threadIdx.x >> 5;
    const int n  = blockIdx.y;
    const int half = lane >> 4, l15 = lane & 15;
    const int p  = (blockIdx.x * 8 + wave) * 16 + l15;
    const int ph = p >> 6, pw = p & 63;
    const bf16_t* xn = x + ((size_t)n << 20);             // 256*4096 elems per sample
    v8f acc[4] = {{}, {}, {}, {}};
    for (int kc = 0; kc < KTOT; kc += 32) {
        v16bf b = {};
#pragma unroll
        for (int e = 0; e < 16; ++e) {
            int k   = kc + 16 * half + e;                 // B element e -> K = 16*halfK + e
            int ci  = k / (KH * KW);
            int rem = k - ci * (KH * KW);
            int kh  = rem / KW;
            int kw  = rem - kh * KW;
            int h = ph + kh - PAD, wc = pw + kw - PAD;
            unsigned short bits = 0;
            if ((unsigned)h < 64u && (unsigned)wc < 64u)
                bits = __builtin_bit_cast(unsigned short, xn[((size_t)ci << 12) + (h << 6) + wc]);
            b[e] = __builtin_bit_cast(__bf16, bits);
        }
#pragma unroll
        for (int t = 0; t < 4; ++t) {
            v16bf a = load_a_rm(w + (size_t)(16 * t + l15) * KTOT + kc, half);
            acc[t] = WMMA_BF16(a, b, acc[t]);
        }
    }
#pragma unroll
    for (int t = 0; t < 4; ++t) {
#pragma unroll
        for (int r = 0; r < 8; ++r) {
            int c = 16 * t + r + 8 * half;                // C/D: VGPR r -> row r+8*half
            float v = acc[t][r] + bias[c];
            size_t idx = TRANS ? (((size_t)n * 4096 + p) * 64 + c)
                               : (((size_t)n * 64 + c) * 4096 + p);
            out[idx] = f2bf(v);
        }
    }
}

// qk[i,j] = sum_c k[c,i]*q[c,j].  A = kT (i,c), B = qT (j,c), K=64.
// One wave computes a 16(i) x 64(j) strip: A fragment loaded once, 4 B tiles.
// Stores qk transposed (j,i) as bf16 + fused per-sample running max.
__global__ void qk_gemm_kernel(const bf16_t* __restrict__ kT, const bf16_t* __restrict__ qT,
                               bf16_t* __restrict__ qk, unsigned* __restrict__ dmax) {
    const int lane = threadIdx.x & 31, wave = threadIdx.x >> 5;
    const int n  = blockIdx.z;
    const int i0 = (blockIdx.y * 8 + wave) * 16;
    const int j0 = blockIdx.x * 64;
    const int half = lane >> 4, l15 = lane & 15;
    const bf16_t* kTn = kT + (size_t)n * 4096 * 64;
    const bf16_t* qTn = qT + (size_t)n * 4096 * 64;
    v8f acc[4] = {{}, {}, {}, {}};
#pragma unroll
    for (int kc = 0; kc < 64; kc += 32) {
        v16bf a = load_a_rm(kTn + (size_t)(i0 + l15) * 64 + kc, half);
#pragma unroll
        for (int t = 0; t < 4; ++t) {
            v16bf b = load_b_tr(qTn + (size_t)(j0 + 16 * t + l15) * 64 + kc, half);
            acc[t] = WMMA_BF16(a, b, acc[t]);
        }
    }
    float m = -3.4e38f;
#pragma unroll
    for (int t = 0; t < 4; ++t) {
        size_t base = ((size_t)n << 24) + (size_t)(j0 + 16 * t + l15) * 4096 + i0;
#pragma unroll
        for (int r = 0; r < 8; ++r) {
            float v = acc[t][r];
            m = fmaxf(m, v);
            qk[base + r + 8 * half] = f2bf(v);
        }
    }
    for (int off = 16; off > 0; off >>= 1) m = fmaxf(m, __shfl_xor(m, off));
    if (lane == 0) atomicMax(dmax + n, flipf(m));
}

// p = exp(qk - max[n]); accumulate per-sample sum. 1<<24 elements per sample.
__global__ void softmax_exp_kernel(const bf16_t* __restrict__ qk, bf16_t* __restrict__ p,
                                   const unsigned* __restrict__ dmax, float* __restrict__ dsum) {
    size_t idx = (size_t)blockIdx.x * 256 + threadIdx.x;
    int n = (int)(idx >> 24);
    float e = __expf(bf2f(qk[idx]) - unflipf(dmax[n]));
    p[idx] = f2bf(e);
    __shared__ float red[256];
    red[threadIdx.x] = e;
    __syncthreads();
    for (int s = 128; s > 0; s >>= 1) {
        if (threadIdx.x < (unsigned)s) red[threadIdx.x] += red[threadIdx.x + s];
        __syncthreads();
    }
    if (threadIdx.x == 0) atomicAdd(dsum + n, red[0]);
}

// att[c,j] = (1/sum) * sum_i v[c,i]*p[i,j].  A = v (c,i), B = pT (j,i), K=4096.
// One wave computes the full 64(c) x 16(j) strip: the huge B matrix (p) is
// streamed exactly once; A (v) is tiny and L2-resident.
// Stores att transposed (j,c) bf16.
__global__ void att_gemm_kernel(const bf16_t* __restrict__ v, const bf16_t* __restrict__ p,
                                const float* __restrict__ dsum, bf16_t* __restrict__ att_t) {
    const int lane = threadIdx.x & 31, wave = threadIdx.x >> 5;
    const int n  = blockIdx.y;
    const int j0 = (blockIdx.x * 8 + wave) * 16;
    const int half = lane >> 4, l15 = lane & 15;
    const bf16_t* vn   = v + (size_t)n * 64 * 4096;
    const bf16_t* brow = p + ((size_t)n << 24) + (size_t)(j0 + l15) * 4096;
    v8f acc[4] = {{}, {}, {}, {}};
    for (int kc = 0; kc < 4096; kc += 32) {
        v16bf b = load_b_tr(brow + kc, half);
#pragma unroll
        for (int t = 0; t < 4; ++t) {
            v16bf a = load_a_rm(vn + (size_t)(16 * t + l15) * 4096 + kc, half);
            acc[t] = WMMA_BF16(a, b, acc[t]);
        }
    }
    float inv = 1.0f / dsum[n];
    size_t base = ((size_t)n * 4096 + j0 + l15) * 64;
#pragma unroll
    for (int t = 0; t < 4; ++t)
#pragma unroll
        for (int r = 0; r < 8; ++r)
            att_t[base + 16 * t + r + 8 * half] = f2bf(acc[t][r] * inv);
}

// out[co,j] = g*(sum_c watt[co,c]*att[c,j] + batt[co]) + g.  A = watt (co,c), B = att_t (j,c).
__global__ void out_gemm_kernel(const bf16_t* __restrict__ watt, const bf16_t* __restrict__ att_t,
                                const float* __restrict__ batt, const float* __restrict__ g,
                                float* __restrict__ out) {
    const int lane = threadIdx.x & 31, wave = threadIdx.x >> 5;
    const int n   = blockIdx.z;
    const int co0 = (blockIdx.y * 8 + wave) * 16;
    const int j0  = blockIdx.x * 16;
    const int half = lane >> 4, l15 = lane & 15;
    const bf16_t* arow = watt + (size_t)(co0 + l15) * 64;
    const bf16_t* brow = att_t + ((size_t)n * 4096 + j0 + l15) * 64;
    v8f acc = {};
#pragma unroll
    for (int kc = 0; kc < 64; kc += 32) {
        v16bf a = load_a_rm(arow + kc, half);
        v16bf b = load_b_tr(brow + kc, half);
        acc = WMMA_BF16(a, b, acc);
    }
    const int j = j0 + l15;
#pragma unroll
    for (int r = 0; r < 8; ++r) {
        int co = co0 + r + 8 * half;
        size_t idx = (((size_t)n * 256 + co) << 12) + j;
        float gv = g[idx];
        out[idx] = gv * (acc[r] + batt[co]) + gv;
    }
}

extern "C" void kernel_launch(void* const* d_in, const int* in_sizes, int n_in,
                              void* d_out, int out_size, void* d_ws, size_t ws_size,
                              hipStream_t stream) {
    const float* x    = (const float*)d_in[0];
    const float* g    = (const float*)d_in[1];
    const float* wq   = (const float*)d_in[2];
    const float* bq   = (const float*)d_in[3];
    const float* wk   = (const float*)d_in[4];
    const float* bk   = (const float*)d_in[5];
    const float* wv   = (const float*)d_in[6];
    const float* bv   = (const float*)d_in[7];
    const float* watt = (const float*)d_in[8];
    const float* batt = (const float*)d_in[9];
    float* out = (float*)d_out;

    char* ws = (char*)d_ws;
    size_t off = 0;
    auto alloc = [&](size_t bytes) -> void* {
        void* p = ws + off;
        off = (off + bytes + 255) & ~(size_t)255;
        return p;
    };
    const size_t NX   = 8ull * 256 * 4096;   // feature map elements
    const size_t NQKV = 8ull * 64 * 4096;    // q/k/v elements
    const size_t NQK  = 8ull << 24;          // 8 * 4096 * 4096

    bf16_t* x_bf    = (bf16_t*)alloc(2 * NX);
    bf16_t* wq_bf   = (bf16_t*)alloc(2ull * 64 * 2304);
    bf16_t* wk_bf   = (bf16_t*)alloc(2ull * 64 * 6400);
    bf16_t* wv_bf   = (bf16_t*)alloc(2ull * 64 * 12544);
    bf16_t* watt_bf = (bf16_t*)alloc(2ull * 256 * 64);
    bf16_t* qT      = (bf16_t*)alloc(2 * NQKV);
    bf16_t* kT      = (bf16_t*)alloc(2 * NQKV);
    bf16_t* vb      = (bf16_t*)alloc(2 * NQKV);
    bf16_t* att_t   = (bf16_t*)alloc(2 * NQKV);
    bf16_t* qk_bf   = (bf16_t*)alloc(2 * NQK);
    bf16_t* p_bf    = (bf16_t*)alloc(2 * NQK);
    unsigned* dmax  = (unsigned*)alloc(8 * sizeof(unsigned));
    float*    dsum  = (float*)alloc(8 * sizeof(float));

    auto cvt = [&](const float* src, bf16_t* dst, size_t n) {
        cvt_f32_bf16_kernel<<<(unsigned)((n + 255) / 256), 256, 0, stream>>>(src, dst, (int)n);
    };
    cvt(x, x_bf, NX);
    cvt(wq, wq_bf, 64ull * 2304);
    cvt(wk, wk_bf, 64ull * 6400);
    cvt(wv, wv_bf, 64ull * 12544);
    cvt(watt, watt_bf, 256ull * 64);

    conv_wmma_kernel<3, 3, 1, true ><<<dim3(32, 8), 256, 0, stream>>>(x_bf, wq_bf, bq, qT);
    conv_wmma_kernel<5, 5, 2, true ><<<dim3(32, 8), 256, 0, stream>>>(x_bf, wk_bf, bk, kT);
    conv_wmma_kernel<7, 7, 3, false><<<dim3(32, 8), 256, 0, stream>>>(x_bf, wv_bf, bv, vb);

    init_stats_kernel<<<1, 32, 0, stream>>>(dmax, dsum);
    qk_gemm_kernel<<<dim3(64, 32, 8), 256, 0, stream>>>(kT, qT, qk_bf, dmax);
    softmax_exp_kernel<<<(unsigned)(NQK / 256), 256, 0, stream>>>(qk_bf, p_bf, dmax, dsum);
    att_gemm_kernel<<<dim3(32, 8), 256, 0, stream>>>(vb, p_bf, dsum, att_t);
    out_gemm_kernel<<<dim3(256, 2, 8), 256, 0, stream>>>(watt_bf, att_t, batt, g, out);
}
